// fasNet_16441134809220
// MI455X (gfx1250) — compile-verified, matrix operand order checked by reading, not run on hardware
//
#include <hip/hip_runtime.h>
#include <hip/hip_bf16.h>
#include <math.h>

// ---------------------------------------------------------------------------
// fasNet (VGG16-style) forward for MI455X / gfx1250 (wave32, WMMA).
//  - convs: implicit GEMM, v_wmma_f32_16x16x32_bf16; k-major weights; B tiles
//    double-buffered into LDS with global_load_async_to_lds_b128 (ASYNCcnt)
//    when available, ds_store_b128 fallback otherwise; branchless A loads.
//  - FCs: split-K v_wmma_f32_16x16x4_f32 partials (HBM-bound: ~6.6 GB weights
//    streamed once, nontemporal), then a reduce kernel with bias+activation.
// ---------------------------------------------------------------------------

typedef __attribute__((ext_vector_type(16))) __bf16 v16bf;
typedef __attribute__((ext_vector_type(8)))  float  v8f;
typedef __attribute__((ext_vector_type(2)))  float  v2f;

#if defined(__has_builtin)
#if __has_builtin(__builtin_amdgcn_global_load_async_to_lds_b128) && \
    __has_builtin(__builtin_amdgcn_s_wait_asynccnt)
#define USE_ASYNC_LDS 1
#endif
#endif
#ifndef USE_ASYNC_LDS
#define USE_ASYNC_LDS 0
#endif

// ---------------------------------------------------------------------------
// Weight transform: wT[(r*Cin+ci)*Cout + n] = w[n*Cin*9 + ci*9 + r]
// ---------------------------------------------------------------------------
__global__ void transpose_w(const float* __restrict__ w, float* __restrict__ wT,
                            int Cin, int Cout)
{
    const int K9 = Cin * 9;
    const long total = (long)K9 * Cout;
    long idx = (long)blockIdx.x * blockDim.x + threadIdx.x;
    if (idx >= total) return;
    const int n = idx % Cout;
    const int k = (int)(idx / Cout);
    const int r = k / Cin;
    const int ci = k - r * Cin;
    wT[idx] = w[(size_t)n * K9 + ci * 9 + r];
}

// ---------------------------------------------------------------------------
// B-tile staging: 32 k-rows x 16 n-cols of f32 into LDS. Lane = k-row.
// ---------------------------------------------------------------------------
#if USE_ASYNC_LDS
typedef __attribute__((vector_size(16))) int vi4;
typedef __attribute__((address_space(1))) vi4 gvi4;
typedef __attribute__((address_space(3))) vi4 lvi4;

__device__ __forceinline__
void stage_B_async(float (*t)[16], const float* __restrict__ wT,
                   int c0, int K9, int Cout, int n0, int lane)
{
    const int kk  = c0 + lane;
    const int kks = (kk < K9) ? kk : (K9 - 1);
    const float* wrow = wT + (size_t)kks * Cout + n0;
#pragma unroll
    for (int j = 0; j < 4; ++j) {
        gvi4* src = (gvi4*)(uintptr_t)(wrow + 4 * j);
        lvi4* dst = (lvi4*)(unsigned int)(uintptr_t)(&t[lane][4 * j]);
        __builtin_amdgcn_global_load_async_to_lds_b128(src, dst, 0, 0);
    }
}
#endif

__device__ __forceinline__
void stage_B_sync(float (*t)[16], const float* __restrict__ wT,
                  int c0, int K9, int Cout, int n0, int lane)
{
    const int kk  = c0 + lane;
    const int kks = (kk < K9) ? kk : (K9 - 1);
    const float* wrow = wT + (size_t)kks * Cout + n0;
#pragma unroll
    for (int j = 0; j < 4; ++j)
        *(float4*)&t[lane][4 * j] = *(const float4*)(wrow + 4 * j);
}

// Fragment build from the f32 LDS tile (bf16 cvt here; k>=K9 masked to zero).
__device__ __forceinline__
v16bf build_Bfrag(const float (*t)[16], int c0, int K9, int mn, int half8)
{
    v16bf b;
#pragma unroll
    for (int e = 0; e < 16; ++e) {
        const int koff = e + ((e < 8) ? 0 : 8) + half8;
        const float v = t[koff][mn];
        b[e] = (c0 + koff < K9) ? (__bf16)v : (__bf16)0.0f;
    }
    return b;
}

// ---------------------------------------------------------------------------
// Conv 3x3 SAME + bias + ReLU. One wave per (batch, 16-ochans, 16-pixel tile).
// K enumerated r-major (tap outer, ci inner) to match wT.
// ---------------------------------------------------------------------------
__global__ __launch_bounds__(32)
void conv3x3_wmma(const float* __restrict__ x, const float* __restrict__ wT,
                  const float* __restrict__ bias, float* __restrict__ y,
                  int B, int Cin, int H, int W, int Cout)
{
    __shared__ __attribute__((aligned(16))) float ldsBF[2][32][16];

    const int lane = threadIdx.x & 31;
    const int HW = H * W;
    const int ntiles = (HW + 15) >> 4;

    int bid = blockIdx.x;
    const int tile = bid % ntiles;  bid /= ntiles;
    const int og   = bid % (Cout >> 4);  bid /= (Cout >> 4);
    const int b    = bid;

    const int n0    = og << 4;
    const int K9    = Cin * 9;
    const int half  = lane >> 4;
    const int half8 = half << 3;
    const int mn    = lane & 15;

    const int p       = tile * 16 + mn;
    const bool validm = (p < HW);
    const int pc = validm ? p : 0;
    const int oh = pc / W;
    const int ow = pc - oh * W;

    const size_t xbase = (size_t)b * Cin * HW;

    v8f acc = {};

    if (Cin >= 32) {
        const int cpr = Cin >> 5;                  // chunks per tap (power of 2)
        const int shp = 31 - __builtin_clz(cpr);
        const int nch = 9 * cpr;

#if USE_ASYNC_LDS
        stage_B_async(ldsBF[0], wT, 0, K9, Cout, n0, lane);
        int cur = 0;
#endif
        for (int ch = 0; ch < nch; ++ch) {
#if USE_ASYNC_LDS
            const bool more = (ch + 1 < nch);
            if (more)
                stage_B_async(ldsBF[cur ^ 1], wT, (ch + 1) << 5, K9, Cout, n0, lane);
            if (more) __builtin_amdgcn_s_wait_asynccnt(4);   // retire older group
            else      __builtin_amdgcn_s_wait_asynccnt(0);
            __syncthreads();
            const float (*bt)[16] = ldsBF[cur];
#else
            stage_B_sync(ldsBF[0], wT, ch << 5, K9, Cout, n0, lane);
            __syncthreads();
            const float (*bt)[16] = ldsBF[0];
#endif
            const int c0 = ch << 5;
            const v16bf bfrag = build_Bfrag(bt, c0, K9, mn, half8);

            // tap geometry for this chunk (uniform-cheap, no per-element div)
            const int r   = ch >> shp;
            const int cb  = (ch - (r << shp)) << 5;
            const int kh  = r / 3;
            const int kw  = r - kh * 3;
            const int ih  = oh + kh - 1;
            const int iw  = ow + kw - 1;
            const bool amask = validm & (ih >= 0) & (ih < H) & (iw >= 0) & (iw < W);
            const int pd = amask ? (ih * W + iw) : 0;

            v16bf afrag;
#pragma unroll
            for (int e = 0; e < 16; ++e) {
                const int koff = e + ((e < 8) ? 0 : 8) + half8;
                const float v = x[xbase + (size_t)(cb + koff) * HW + pd];
                afrag[e] = amask ? (__bf16)v : (__bf16)0.0f;
            }
            acc = __builtin_amdgcn_wmma_f32_16x16x32_bf16(
                      false, afrag, false, bfrag, (short)0, acc, false, false);
            __syncthreads();
#if USE_ASYNC_LDS
            cur ^= 1;
#endif
        }
    } else {
        // generic path (layer 0, Cin=3): per-element tap decode
        for (int c0 = 0; c0 < K9; c0 += 32) {
            stage_B_sync(ldsBF[0], wT, c0, K9, Cout, n0, lane);
            __syncthreads();
            const v16bf bfrag = build_Bfrag(ldsBF[0], c0, K9, mn, half8);

            v16bf afrag;
#pragma unroll
            for (int e = 0; e < 16; ++e) {
                const int koff = e + ((e < 8) ? 0 : 8) + half8;
                const int k = c0 + koff;
                const bool kok = (k < K9);
                const int ks = kok ? k : 0;
                const int r  = ks / Cin;
                const int ci = ks - r * Cin;
                const int kh = r / 3;
                const int kw = r - kh * 3;
                const int ih = oh + kh - 1;
                const int iw = ow + kw - 1;
                const bool amask =
                    kok & validm & (ih >= 0) & (ih < H) & (iw >= 0) & (iw < W);
                const int pd = amask ? (ih * W + iw) : 0;
                const float v = x[xbase + (size_t)ci * HW + pd];
                afrag[e] = amask ? (__bf16)v : (__bf16)0.0f;
            }
            acc = __builtin_amdgcn_wmma_f32_16x16x32_bf16(
                      false, afrag, false, bfrag, (short)0, acc, false, false);
            __syncthreads();
        }
    }

    // D layout: n = lane&15 ; m = vgpr + 8*half
    const int nn = n0 + mn;
    const float bb = bias[nn];
#pragma unroll
    for (int r = 0; r < 8; ++r) {
        const int m  = r + half8;
        const int pp = tile * 16 + m;
        if (pp < HW) {
            float v = acc[r] + bb;
            y[((size_t)(b * Cout + nn)) * HW + pp] = fmaxf(v, 0.0f);
        }
    }
}

// ---------------------------------------------------------------------------
// FC split-K partial: part[seg][m][n] = x[m, kbeg:kend] @ W[kbeg:kend, n]
// ---------------------------------------------------------------------------
__global__ __launch_bounds__(32)
void fc_partial(const float* __restrict__ x, const float* __restrict__ w,
                float* __restrict__ part, int M, int I, int O, int KS, int S)
{
    const int lane = threadIdx.x & 31;
    const int half = lane >> 4;
    const int mn   = lane & 15;

    int bid = blockIdx.x;
    const int seg = bid % S;
    const int og  = bid / S;
    const int n0  = og * 16;
    const int n   = n0 + mn;
    const bool nvalid = (n < O);
    const int nc = nvalid ? n : (O - 1);

    const bool mrow = (mn < M);
    const size_t xrow = (size_t)(mrow ? mn : 0) * I;

    const int kbeg = seg * KS;
    const int kend = min(I, kbeg + KS);   // I, KS multiples of 16 -> no ragged tail

    v8f acc = {};
    for (int k0 = kbeg; k0 < kend; k0 += 16) {
#pragma unroll
        for (int u = 0; u < 4; ++u) {
            const int ka = k0 + 4 * u + 2 * half;
            v2f a, bf;
            const float a0 = x[xrow + ka];
            const float a1 = x[xrow + ka + 1];
            a.x = mrow ? a0 : 0.0f;
            a.y = mrow ? a1 : 0.0f;
            bf.x = __builtin_nontemporal_load(&w[(size_t)ka * O + nc]);
            bf.y = __builtin_nontemporal_load(&w[(size_t)(ka + 1) * O + nc]);
            acc = __builtin_amdgcn_wmma_f32_16x16x4_f32(
                      false, a, false, bf, (short)0, acc, false, false);
        }
    }

#pragma unroll
    for (int r = 0; r < 8; ++r) {
        const int m = r + 8 * half;
        if (m < M && nvalid)
            part[((size_t)seg * M + m) * O + n] = acc[r];
    }
}

__global__ void fc_reduce(const float* __restrict__ part, const float* __restrict__ bias,
                          float* __restrict__ y, int M, int O, int S, int act)
{
    const long total = (long)M * O;
    long idx = (long)blockIdx.x * blockDim.x + threadIdx.x;
    if (idx >= total) return;
    const int n = idx % O;
    const int m = (int)(idx / O);
    float v = bias[n];
    for (int s = 0; s < S; ++s)
        v += part[((size_t)s * M + m) * O + n];
    if (act == 1)      v = fmaxf(v, 0.0f);
    else if (act == 2) v = 1.0f / (1.0f + __expf(-v));
    y[(size_t)m * O + n] = v;
}

// ---------------------------------------------------------------------------
// Elementwise helpers
// ---------------------------------------------------------------------------
__global__ void maxpool2x2(const float* __restrict__ x, float* __restrict__ y,
                           int BC, int H, int W)
{
    const int oH = H >> 1, oW = W >> 1;
    const long total = (long)BC * oH * oW;
    long idx = (long)blockIdx.x * blockDim.x + threadIdx.x;
    if (idx >= total) return;
    int ow = idx % oW; long t = idx / oW;
    int oh = t % oH;   int bc = (int)(t / oH);
    const float* p = x + ((size_t)bc * H + (size_t)oh * 2) * W + (size_t)ow * 2;
    y[idx] = fmaxf(fmaxf(p[0], p[1]), fmaxf(p[W], p[W + 1]));
}

__global__ void landmark_post(const float* __restrict__ lm_raw,
                              float* __restrict__ out_sig, int* __restrict__ starts)
{
    const int i = threadIdx.x;           // 32 = [B=2][8 lm][2 coords]
    if (i < 32) {
        const float v = lm_raw[i];
        out_sig[i] = 1.0f / (1.0f + __expf(-v));
        int st = (int)truncf(v * 14.0f) - 3;
        st = st < 0 ? 0 : (st > 7 ? 7 : st);
        starts[i] = st;
    }
}

__global__ void crop_local(const float* __restrict__ feat,
                           const int* __restrict__ starts,
                           float* __restrict__ local, int B)
{
    const long total = (long)B * 8 * 512 * 49;
    long idx = (long)blockIdx.x * blockDim.x + threadIdx.x;
    if (idx >= total) return;
    int col = idx % 7;  long t = idx / 7;
    int row = t % 7;    t /= 7;
    int c   = t % 512;  t /= 512;
    int j   = t % 8;    int b = (int)(t / 8);
    const int sr = starts[b * 16 + j * 2 + 0];
    const int sc = starts[b * 16 + j * 2 + 1];
    local[idx] = feat[(((size_t)b * 512 + c) * 14 + sr + row) * 14 + sc + col];
}

__global__ void concat2(const float* __restrict__ a, const float* __restrict__ b,
                        float* __restrict__ y, int M, int Na, int Nb)
{
    const long total = (long)M * (Na + Nb);
    long idx = (long)blockIdx.x * blockDim.x + threadIdx.x;
    if (idx >= total) return;
    const int n = idx % (Na + Nb);
    const int m = (int)(idx / (Na + Nb));
    y[idx] = (n < Na) ? a[(size_t)m * Na + n] : b[(size_t)m * Nb + (n - Na)];
}

__global__ void copyf4(const float* __restrict__ s, float* __restrict__ d, long n4)
{
    long idx = (long)blockIdx.x * blockDim.x + threadIdx.x;
    if (idx < n4) ((float4*)d)[idx] = ((const float4*)s)[idx];
}

// ---------------------------------------------------------------------------
// Host side
// ---------------------------------------------------------------------------
static inline void launch_conv(const float* in, const float* w, const float* b,
                               float* wT, float* out, int Bn, int Cin, int H, int W,
                               int Cout, hipStream_t s)
{
    const long wtot = (long)Cin * 9 * Cout;
    transpose_w<<<(int)((wtot + 255) / 256), 256, 0, s>>>(w, wT, Cin, Cout);
    const int HW = H * W;
    const int ntiles = (HW + 15) / 16;
    const int grid = Bn * (Cout / 16) * ntiles;
    conv3x3_wmma<<<grid, 32, 0, s>>>(in, wT, b, out, Bn, Cin, H, W, Cout);
}

static inline void launch_fc(const float* x, const float* w, const float* b,
                             float* y, float* part, int M, int I, int O, int act,
                             hipStream_t s)
{
    const int KS = 4096;
    const int S = (I + KS - 1) / KS;
    const int ntiles = (O + 15) / 16;
    fc_partial<<<ntiles * S, 32, 0, s>>>(x, w, part, M, I, O, KS, S);
    const long tot = (long)M * O;
    fc_reduce<<<(int)((tot + 255) / 256), 256, 0, s>>>(part, b, y, M, O, S, act);
}

static inline void launch_pool(const float* x, float* y, int BC, int H, int W,
                               hipStream_t s)
{
    const long total = (long)BC * (H / 2) * (W / 2);
    maxpool2x2<<<(int)((total + 255) / 256), 256, 0, s>>>(x, y, BC, H, W);
}

extern "C" void kernel_launch(void* const* d_in, const int* in_sizes, int n_in,
                              void* d_out, int out_size, void* d_ws, size_t ws_size,
                              hipStream_t stream)
{
    (void)in_sizes; (void)n_in; (void)out_size; (void)ws_size;
    const int B = 2;

    const float* x = (const float*)d_in[0];
    auto P = [&](int i) { return (const float*)d_in[i]; };
    const float *fw[10], *fb[10];
    for (int i = 0; i < 10; ++i) { fw[i] = P(1 + 2 * i); fb[i] = P(2 + 2 * i); }
    const float *lcw[3], *lcb[3];
    for (int i = 0; i < 3; ++i)  { lcw[i] = P(21 + 2 * i); lcb[i] = P(22 + 2 * i); }
    const float *lm_w1 = P(27), *lm_b1 = P(28);
    const float *lm_w2 = P(29), *lm_b2 = P(30);
    const float *lm_w3 = P(31), *lm_b3 = P(32);
    const float *wl = P(33), *bl = P(34);
    const float *wg = P(35), *bg = P(36);
    const float *wf1 = P(37), *bf1 = P(38);
    const float *wf2 = P(39), *bf2 = P(40);

    // ---- workspace layout (floats) ----
    float* ws = (float*)d_ws;
    float* bufA = ws;                       // 6,422,528
    float* bufB = bufA + 6422528;           // 6,422,528
    float* feat = bufB + 6422528;           // 200,704
    float* h0   = feat + 200704;            // 200,704
    float* h1   = h0 + 200704;              // 200,704
    float* lmh1 = h1 + 200704;              // 8,192
    float* lmh2 = lmh1 + 8192;              // 8,192
    float* lmr  = lmh2 + 8192;              // 32
    float* loc  = lmr + 32;                 // 401,408
    float* lfc  = loc + 401408;             // 8,192
    float* gfc  = lfc + 8192;               // 8,192
    float* comb = gfc + 8192;               // 16,384
    float* c1   = comb + 16384;             // 8,192
    float* wT   = c1 + 8192;                // 2,359,296 (max conv wT: 512*4608)
    float* part = wT + 2359296;             // 524,288 (max split-K partials)
    int*   st   = (int*)(part + 524288);    // 32 ints

    float* out = (float*)d_out;             // [32 | 200704 | 2100]

    // ---- VGG features ----
    launch_conv(x,    fw[0], fb[0], wT, bufA, B,   3, 224, 224,  64, stream);
    launch_conv(bufA, fw[1], fb[1], wT, bufB, B,  64, 224, 224,  64, stream);
    launch_pool(bufB, bufA, B * 64, 224, 224, stream);
    launch_conv(bufA, fw[2], fb[2], wT, bufB, B,  64, 112, 112, 128, stream);
    launch_conv(bufB, fw[3], fb[3], wT, bufA, B, 128, 112, 112, 128, stream);
    launch_pool(bufA, bufB, B * 128, 112, 112, stream);
    launch_conv(bufB, fw[4], fb[4], wT, bufA, B, 128,  56,  56, 256, stream);
    launch_conv(bufA, fw[5], fb[5], wT, bufB, B, 256,  56,  56, 256, stream);
    launch_conv(bufB, fw[6], fb[6], wT, bufA, B, 256,  56,  56, 256, stream);
    launch_pool(bufA, bufB, B * 256, 56, 56, stream);
    launch_conv(bufB, fw[7], fb[7], wT, bufA, B, 256,  28,  28, 512, stream);
    launch_conv(bufA, fw[8], fb[8], wT, bufB, B, 512,  28,  28, 512, stream);
    launch_conv(bufB, fw[9], fb[9], wT, bufA, B, 512,  28,  28, 512, stream);
    launch_pool(bufA, feat, B * 512, 28, 28, stream);

    // ---- landmark branch ----
    launch_conv(feat, lcw[0], lcb[0], wT, h0, B, 512, 14, 14, 512, stream);
    launch_conv(h0,   lcw[1], lcb[1], wT, h1, B, 512, 14, 14, 512, stream);
    launch_conv(h1,   lcw[2], lcb[2], wT, h0, B, 512, 14, 14, 512, stream);

    launch_fc(h0,   lm_w1, lm_b1, lmh1, part, B, 512 * 14 * 14, 4096, 1, stream);
    launch_fc(lmh1, lm_w2, lm_b2, lmh2, part, B, 4096,          4096, 1, stream);
    launch_fc(lmh2, lm_w3, lm_b3, lmr,  part, B, 4096,            16, 0, stream);

    landmark_post<<<1, 32, 0, stream>>>(lmr, out + 0, st);

    {
        const long total = (long)B * 8 * 512 * 49;
        crop_local<<<(int)((total + 255) / 256), 256, 0, stream>>>(feat, st, loc, B);
    }
    launch_fc(loc,  wl, bl, lfc, part, B, 8 * 512 * 49,  4096, 1, stream);
    launch_fc(feat, wg, bg, gfc, part, B, 512 * 14 * 14, 4096, 1, stream);

    {
        const long total = (long)B * 8192;
        concat2<<<(int)((total + 255) / 256), 256, 0, stream>>>(lfc, gfc, comb, B, 4096, 4096);
    }

    launch_fc(comb, wf1, bf1, c1, part, B, 8192, 4096, 1, stream);
    launch_fc(c1,   wf2, bf2, out + 32 + 200704, part, B, 4096, 1050, 2, stream);

    copyf4<<<(200704 / 4 + 255) / 256, 256, 0, stream>>>(feat, out + 32, 200704 / 4);
}
